// DecorConv_52828097741062
// MI455X (gfx1250) — matrix operand order repaired and checked by reading, not compiled, not run on hardware
//
#include <hip/hip_runtime.h>
#include <hip/hip_bf16.h>
#include <math.h>

// ---------------------------------------------------------------------------
// DecorConv on MI455X (gfx1250, wave32, WMMA):
//   d[b]   = decor^T @ im2col(input[b])        (bf16 WMMA, f32 accum)
//   norm_i = sqrt(mean u_i^2)/(sqrt(mean d_i^2)+1e-8)
//   out[b] = (conv * norm) @ d[b]              (bf16 WMMA, f32 accum)
// d kept in bf16 in d_ws (75 MB); input pre-padded to bf16 (34x34) so the
// fused im2col is a single contiguous 32B read per thread per K-step.
// Both GEMMs are software-pipelined (global->regs overlapped with WMMA).
// ---------------------------------------------------------------------------

typedef __attribute__((ext_vector_type(16))) __bf16 v16bf;
typedef __attribute__((ext_vector_type(8)))  float  v8f;

union Frag { v16bf v; unsigned u[8]; };
union BReg { uint4 q[2]; unsigned short s[16]; };

__device__ __forceinline__ unsigned short f2bf(float f) {
  unsigned u = __builtin_bit_cast(unsigned, f);
  u += 0x7FFFu + ((u >> 16) & 1u);          // round-to-nearest-even
  return (unsigned short)(u >> 16);
}

#define NFEAT 576
#define LPIX  1024
#define NB    64
#define NOUT  256
#define MT    64
#define NT    128
#define KT    32
#define PW    34                           // padded width/height
#define PPLANE (PW * PW)                   // 1156

// ---------------- prep: At[m][k] = bf16(decor[k][m]) ----------------------
__global__ void prep_at_kernel(const float* __restrict__ decor,
                               unsigned short* __restrict__ At) {
  int idx = blockIdx.x * 256 + threadIdx.x;
  if (idx < NFEAT * NFEAT) {
    int m = idx / NFEAT, k = idx - m * NFEAT;
    At[idx] = f2bf(decor[k * NFEAT + m]);
  }
}

// ---------------- prep: zero-padded bf16 input (B,C,34,34) ----------------
__global__ void pad_input_kernel(const float* __restrict__ inp,
                                 unsigned short* __restrict__ pad) {
  int idx = blockIdx.x * 256 + threadIdx.x;
  if (idx < NB * 64 * PPLANE) {
    const int x  = idx % PW;
    const int t1 = idx / PW;
    const int y  = t1 % PW;
    const int bc = t1 / PW;                // b*64 + c
    unsigned short v = 0;
    if ((unsigned)(y - 1) < 32u && (unsigned)(x - 1) < 32u)
      v = f2bf(inp[(size_t)bc * 1024 + (y - 1) * 32 + (x - 1)]);
    pad[idx] = v;
  }
}

// ---------------- ssq_u[nf] = sum_{b,l} u^2  (from f32 input) -------------
__global__ void ssq_u_kernel(const float* __restrict__ inp,
                             float* __restrict__ ssq_u) {
  const int nf = blockIdx.x;                 // 0..575
  const int c  = nf / 9;
  const int r9 = nf - c * 9;
  const int kh = r9 / 3, kw = r9 - kh * 3;
  float s = 0.f;
  for (int idx = threadIdx.x; idx < NB * LPIX; idx += 256) {
    const int b = idx >> 10;
    const int h = (idx >> 5) & 31;
    const int w = idx & 31;
    const int ih = h + kh - 1, iw = w + kw - 1;
    if ((unsigned)ih < 32u && (unsigned)iw < 32u) {
      const float v = inp[((size_t)b * 64 + c) * 1024 + ih * 32 + iw];
      s += v * v;
    }
  }
  __shared__ float red[256];
  red[threadIdx.x] = s;
  __syncthreads();
  for (int st = 128; st > 0; st >>= 1) {
    if (threadIdx.x < st) red[threadIdx.x] += red[threadIdx.x + st];
    __syncthreads();
  }
  if (threadIdx.x == 0) ssq_u[nf] = red[0];
}

// ---------------- GEMM 1: d = At @ im2col(pad), + ssq_d -------------------
__global__ __launch_bounds__(256)
void decor_gemm_kernel(const unsigned short* __restrict__ pad,
                       const unsigned short* __restrict__ At,
                       unsigned short* __restrict__ dbf,
                       float* __restrict__ ssq_d) {
  __shared__ unsigned AsU[MT * 17];          // packed bf16 pairs, stride 17
  __shared__ unsigned BsU[(KT / 2) * NT];    // packed (k,k+1) pairs

  const int mBase = blockIdx.x * MT;
  const int lBase = blockIdx.y * NT;
  const int b     = blockIdx.z;

  const int tid  = threadIdx.x;
  const int lane = tid & 31;
  const int wave = tid >> 5;
  const int mw   = (wave >> 2) * 32;         // 0 / 32
  const int nw   = (wave & 3) * 32;          // 0..96
  const int half = lane >> 4;
  const int col  = lane & 15;

  v8f acc[2][2];
  #pragma unroll
  for (int i = 0; i < 2; ++i)
    #pragma unroll
    for (int j = 0; j < 2; ++j) acc[i][j] = (v8f)0.f;

  const int am = tid >> 2;                   // 0..63
  const int ak = (tid & 3) * 8;              // 0,8,16,24
  const int bk = tid >> 3;                   // 0..31
  const int bl = (tid & 7) * 16;             // 0..112
  const int kp  = bk >> 1;
  const int klo = bk & 1;

  const int h0 = (lBase + bl) >> 5;          // image row of this 16-chunk
  const int w0 = (lBase + bl) & 31;
  const unsigned short* Pb = pad + (size_t)b * 64 * PPLANE;
  const unsigned short* aRow = At + (size_t)(mBase + am) * NFEAT + ak;
  unsigned short* bs16 = reinterpret_cast<unsigned short*>(BsU);

  // ---- stage 0 loads
  uint4 aReg = *reinterpret_cast<const uint4*>(aRow);
  BReg bReg;
  {
    const int nf = bk;
    const int c  = nf / 9;
    const int r9 = nf - c * 9;
    const int kh = r9 / 3, kw = r9 - kh * 3;
    __builtin_memcpy(&bReg, Pb + ((size_t)c * PW + h0 + kh) * PW + w0 + kw, 32);
  }
  {
    unsigned* dst = &AsU[am * 17 + (ak >> 1)];
    dst[0] = aReg.x; dst[1] = aReg.y; dst[2] = aReg.z; dst[3] = aReg.w;
    #pragma unroll
    for (int e = 0; e < 16; ++e)
      bs16[((kp * NT + bl + e) << 1) | klo] = bReg.s[e];
  }
  __syncthreads();

  for (int kb = 0; kb < NFEAT; kb += KT) {
    const int kn = kb + KT;
    const bool has = kn < NFEAT;
    // ---- issue next-tile global loads (overlap with WMMA below)
    if (has) {
      aReg = *reinterpret_cast<const uint4*>(aRow + kn);
      const int nf = kn + bk;
      const int c  = nf / 9;
      const int r9 = nf - c * 9;
      const int kh = r9 / 3, kw = r9 - kh * 3;
      __builtin_memcpy(&bReg, Pb + ((size_t)c * PW + h0 + kh) * PW + w0 + kw, 32);
      __builtin_prefetch(aRow + kn + KT, 0, 1);
    }
    // ---- fragments + WMMA from LDS
    Frag aF[2], bF[2];
    #pragma unroll
    for (int mi = 0; mi < 2; ++mi) {
      const unsigned* arow = &AsU[(mw + mi * 16 + col) * 17];
      #pragma unroll
      for (int j = 0; j < 4; ++j) aF[mi].u[j]     = arow[half * 4 + j];
      #pragma unroll
      for (int j = 0; j < 4; ++j) aF[mi].u[4 + j] = arow[8 + half * 4 + j];
    }
    #pragma unroll
    for (int ni = 0; ni < 2; ++ni) {
      const int c0 = nw + ni * 16 + col;
      #pragma unroll
      for (int j = 0; j < 8; ++j) bF[ni].u[j] = BsU[((j >> 2) * 8 + half * 4 + (j & 3)) * NT + c0];
    }
    #pragma unroll
    for (int mi = 0; mi < 2; ++mi)
      #pragma unroll
      for (int ni = 0; ni < 2; ++ni)
        acc[mi][ni] = __builtin_amdgcn_wmma_f32_16x16x32_bf16(
            false, aF[mi].v, false, bF[ni].v, (short)0, acc[mi][ni],
            false, false);
    __syncthreads();                         // everyone done reading LDS
    if (has) {
      unsigned* dst = &AsU[am * 17 + (ak >> 1)];
      dst[0] = aReg.x; dst[1] = aReg.y; dst[2] = aReg.z; dst[3] = aReg.w;
      #pragma unroll
      for (int e = 0; e < 16; ++e)
        bs16[((kp * NT + bl + e) << 1) | klo] = bReg.s[e];
    }
    __syncthreads();                         // LDS refilled
  }

  // ---- epilogue: d -> bf16, per-row sum of squares -> atomic ssq_d
  #pragma unroll
  for (int mi = 0; mi < 2; ++mi) {
    #pragma unroll
    for (int r = 0; r < 8; ++r) {
      const int m = mBase + mw + mi * 16 + half * 8 + r;
      float s = 0.f;
      #pragma unroll
      for (int ni = 0; ni < 2; ++ni) {
        const float v = acc[mi][ni][r];
        const int l = lBase + nw + ni * 16 + col;
        dbf[((size_t)b * NFEAT + m) * LPIX + l] = f2bf(v);
        s += v * v;
      }
      #pragma unroll
      for (int off = 1; off < 16; off <<= 1) s += __shfl_xor(s, off, 32);
      if (col == 0) atomicAdd(&ssq_d[m], s);
    }
  }
}

// ---------------- norm + weight folding ----------------------------------
__global__ void norm_kernel(const float* __restrict__ ssq_u,
                            const float* __restrict__ ssq_d,
                            float* __restrict__ normv) {
  int i = blockIdx.x * 256 + threadIdx.x;
  if (i < NFEAT) {
    const float inv = 1.f / (float)(NB * LPIX);
    normv[i] = sqrtf(ssq_u[i] * inv) / (sqrtf(ssq_d[i] * inv) + 1e-8f);
  }
}

__global__ void scale_w_kernel(const float* __restrict__ conv,
                               const float* __restrict__ normv,
                               unsigned short* __restrict__ Wp) {
  int idx = blockIdx.x * 256 + threadIdx.x;
  if (idx < NOUT * NFEAT) {
    int nf = idx % NFEAT;
    Wp[idx] = f2bf(conv[idx] * normv[nf]);
  }
}

// ---------------- GEMM 2: out = Wp @ d  (f32 output) ----------------------
__global__ __launch_bounds__(256)
void out_gemm_kernel(const unsigned short* __restrict__ Wp,
                     const unsigned short* __restrict__ dbf,
                     float* __restrict__ out) {
  __shared__ unsigned AsU[MT * 17];
  __shared__ unsigned BsU[(KT / 2) * NT];

  const int mBase = blockIdx.x * MT;         // 0..192
  const int lBase = blockIdx.y * NT;
  const int b     = blockIdx.z;

  const int tid  = threadIdx.x;
  const int lane = tid & 31;
  const int wave = tid >> 5;
  const int mw   = (wave >> 2) * 32;
  const int nw   = (wave & 3) * 32;
  const int half = lane >> 4;
  const int col  = lane & 15;

  v8f acc[2][2];
  #pragma unroll
  for (int i = 0; i < 2; ++i)
    #pragma unroll
    for (int j = 0; j < 2; ++j) acc[i][j] = (v8f)0.f;

  const int am = tid >> 2;
  const int ak = (tid & 3) * 8;
  const int bk = tid >> 3;
  const int bl = (tid & 7) * 16;
  const int kp  = bk >> 1;
  const int klo = bk & 1;

  const unsigned short* aRow = Wp + (size_t)(mBase + am) * NFEAT + ak;
  const unsigned short* bRow =
      dbf + ((size_t)b * NFEAT + bk) * LPIX + lBase + bl;
  unsigned short* bs16 = reinterpret_cast<unsigned short*>(BsU);

  uint4 aReg = *reinterpret_cast<const uint4*>(aRow);
  BReg bReg;
  bReg.q[0] = *reinterpret_cast<const uint4*>(bRow);
  bReg.q[1] = *reinterpret_cast<const uint4*>(bRow + 8);
  {
    unsigned* dst = &AsU[am * 17 + (ak >> 1)];
    dst[0] = aReg.x; dst[1] = aReg.y; dst[2] = aReg.z; dst[3] = aReg.w;
    #pragma unroll
    for (int e = 0; e < 16; ++e)
      bs16[((kp * NT + bl + e) << 1) | klo] = bReg.s[e];
  }
  __syncthreads();

  for (int kb = 0; kb < NFEAT; kb += KT) {
    const int kn = kb + KT;
    const bool has = kn < NFEAT;
    if (has) {
      aReg = *reinterpret_cast<const uint4*>(aRow + kn);
      const unsigned short* bn = bRow + (size_t)kn * LPIX;
      bReg.q[0] = *reinterpret_cast<const uint4*>(bn);
      bReg.q[1] = *reinterpret_cast<const uint4*>(bn + 8);
      __builtin_prefetch(bn + (size_t)KT * LPIX, 0, 1);
    }
    Frag aF[2], bF[2];
    #pragma unroll
    for (int mi = 0; mi < 2; ++mi) {
      const unsigned* arow = &AsU[(mw + mi * 16 + col) * 17];
      #pragma unroll
      for (int j = 0; j < 4; ++j) aF[mi].u[j]     = arow[half * 4 + j];
      #pragma unroll
      for (int j = 0; j < 4; ++j) aF[mi].u[4 + j] = arow[8 + half * 4 + j];
    }
    #pragma unroll
    for (int ni = 0; ni < 2; ++ni) {
      const int c0 = nw + ni * 16 + col;
      #pragma unroll
      for (int j = 0; j < 8; ++j) bF[ni].u[j] = BsU[((j >> 2) * 8 + half * 4 + (j & 3)) * NT + c0];
    }
    #pragma unroll
    for (int mi = 0; mi < 2; ++mi)
      #pragma unroll
      for (int ni = 0; ni < 2; ++ni)
        acc[mi][ni] = __builtin_amdgcn_wmma_f32_16x16x32_bf16(
            false, aF[mi].v, false, bF[ni].v, (short)0, acc[mi][ni],
            false, false);
    __syncthreads();
    if (has) {
      unsigned* dst = &AsU[am * 17 + (ak >> 1)];
      dst[0] = aReg.x; dst[1] = aReg.y; dst[2] = aReg.z; dst[3] = aReg.w;
      #pragma unroll
      for (int e = 0; e < 16; ++e)
        bs16[((kp * NT + bl + e) << 1) | klo] = bReg.s[e];
    }
    __syncthreads();
  }

  #pragma unroll
  for (int mi = 0; mi < 2; ++mi)
    #pragma unroll
    for (int r = 0; r < 8; ++r) {
      const int m = mBase + mw + mi * 16 + half * 8 + r;
      #pragma unroll
      for (int ni = 0; ni < 2; ++ni) {
        const int l = lBase + nw + ni * 16 + col;
        out[((size_t)b * NOUT + m) * LPIX + l] = acc[mi][ni][r];
      }
    }
}

// ---------------- host ----------------------------------------------------
extern "C" void kernel_launch(void* const* d_in, const int* in_sizes, int n_in,
                              void* d_out, int out_size, void* d_ws, size_t ws_size,
                              hipStream_t stream) {
  (void)in_sizes; (void)n_in; (void)out_size; (void)ws_size;
  const float* inp   = (const float*)d_in[0];   // (64,64,32,32)
  const float* decor = (const float*)d_in[1];   // (576,576)
  const float* conv  = (const float*)d_in[2];   // (256,576)
  float* out = (float*)d_out;                   // (64,256,32,32)

  char* w = (char*)d_ws;
  size_t off = 0;
  auto take = [&](size_t bytes) {
    void* p = w + off;
    off = (off + bytes + 255) & ~(size_t)255;
    return p;
  };
  unsigned short* dbf = (unsigned short*)take((size_t)NB * NFEAT * LPIX * 2);
  unsigned short* pad = (unsigned short*)take((size_t)NB * 64 * PPLANE * 2);
  unsigned short* At  = (unsigned short*)take((size_t)NFEAT * NFEAT * 2);
  unsigned short* Wp  = (unsigned short*)take((size_t)NOUT * NFEAT * 2);
  float* ssq_u = (float*)take(NFEAT * sizeof(float));
  float* ssq_d = (float*)take(NFEAT * sizeof(float));
  float* normv = (float*)take(NFEAT * sizeof(float));

  prep_at_kernel<<<(NFEAT * NFEAT + 255) / 256, 256, 0, stream>>>(decor, At);
  pad_input_kernel<<<(NB * 64 * PPLANE + 255) / 256, 256, 0, stream>>>(inp, pad);
  ssq_u_kernel<<<NFEAT, 256, 0, stream>>>(inp, ssq_u);
  hipMemsetAsync(ssq_d, 0, NFEAT * sizeof(float), stream);
  decor_gemm_kernel<<<dim3(NFEAT / MT, LPIX / NT, NB), 256, 0, stream>>>(
      pad, At, dbf, ssq_d);
  norm_kernel<<<(NFEAT + 255) / 256, 256, 0, stream>>>(ssq_u, ssq_d, normv);
  scale_w_kernel<<<(NOUT * NFEAT + 255) / 256, 256, 0, stream>>>(conv, normv, Wp);
  out_gemm_kernel<<<dim3(NOUT / MT, LPIX / NT, NB), 256, 0, stream>>>(
      Wp, dbf, out);
}
